// ElectronicEmbedding_30133490549297
// MI455X (gfx1250) — matrix operand order repaired and checked by reading, not compiled, not run on hardware
//
#include <hip/hip_runtime.h>

// ---------- types ----------
typedef __bf16 bf16_t;
typedef __attribute__((ext_vector_type(16))) __bf16 v16bf;
typedef __attribute__((ext_vector_type(8)))  __bf16 v8bf;
typedef __attribute__((ext_vector_type(4)))  __bf16 v4bf;
typedef __attribute__((ext_vector_type(8)))  float  v8f;

#define F 256
#define HP 260   // fp32 h-tile row pitch (floats), padded: breaks bank conflicts
#define TP 264   // bf16 staging row pitch (elems), padded: 528B stride -> conflict-free b128
#define MTILE 128
#define EPS 1e-8f

__device__ __forceinline__ float silu_f(float x) { return x / (1.0f + __expf(-x)); }

__device__ __forceinline__ bf16_t f2bf(float f) {
    union { float f; unsigned u; } v; v.f = f;
    unsigned r = v.u + 0x7FFFu + ((v.u >> 16) & 1u);   // round-to-nearest-even
    union { unsigned short s; bf16_t b; } o; o.s = (unsigned short)(r >> 16);
    return o.b;
}

// ---------- prep: wkq = Wq^T Wk, kb = Wk . bq ----------
__global__ void make_wkq(const float* __restrict__ Wq, const float* __restrict__ bq,
                         const float* __restrict__ Wk, float* __restrict__ wkq,
                         float* __restrict__ kb) {
    int f = threadIdx.x;               // 256 threads
    float s = 0.f;
    for (int g = 0; g < F; ++g) s += Wk[g] * Wq[g * F + f];
    wkq[f] = s;
    if (f == 0) {
        float t = 0.f;
        for (int g = 0; g < F; ++g) t += Wk[g] * bq[g];
        *kb = t;
    }
}

// ---------- prep: weights fp32 -> bf16 ----------
__global__ void cvt_weights(const float* __restrict__ W1, const float* __restrict__ W2,
                            const float* __restrict__ Wo,
                            bf16_t* __restrict__ o1, bf16_t* __restrict__ o2,
                            bf16_t* __restrict__ oo) {
    int i = blockIdx.x * 256 + threadIdx.x;   // grid covers 131072
    if (i < 2 * F * F) { o1[i] = f2bf(W1[i]); o2[i] = f2bf(W2[i]); }
    if (i < F * F)     { oo[i] = f2bf(Wo[i]); }
}

// ---------- per-atom: a[n] = softplus( (wkq.x + kb) * (e/en)/16 ) ----------
__global__ void dot_softplus(const float* __restrict__ x, const float* __restrict__ E,
                             const int* __restrict__ seg, const float* __restrict__ wkq,
                             const float* __restrict__ kbp, float* __restrict__ a_out,
                             int Nat) {
    int atom = blockIdx.x * 8 + (threadIdx.x >> 5);
    int lane = threadIdx.x & 31;
    if (atom >= Nat) return;
    const float4* xr = (const float4*)(x + (size_t)atom * F);
    const float4* wr = (const float4*)wkq;
    float s = 0.f;
    #pragma unroll
    for (int i = 0; i < 2; ++i) {
        int idx = lane + 32 * i;
        float4 xv = xr[idx]; float4 wv = wr[idx];
        s += xv.x * wv.x + xv.y * wv.y + xv.z * wv.z + xv.w * wv.w;
    }
    #pragma unroll
    for (int off = 16; off > 0; off >>= 1) s += __shfl_xor(s, off, 32);
    if (lane == 0) {
        int b = seg[atom];
        float e  = fabsf(E[b]);
        float en = fmaxf(e, 1.0f);
        float d  = (s + *kbp) * (e / en) * 0.0625f;     // 1/sqrt(256)
        a_out[atom] = fmaxf(d, 0.f) + log1pf(expf(-fabsf(d)));  // stable softplus
    }
}

// ---------- deterministic segment sum (batch_seg is sorted) ----------
__global__ void seg_reduce(const float* __restrict__ a, const int* __restrict__ seg,
                           float* __restrict__ anorm, int Nat) {
    int b = blockIdx.x;
    int lo = 0, hi = Nat;
    while (lo < hi) { int m = (lo + hi) >> 1; if (seg[m] < b) lo = m + 1; else hi = m; }
    int s0 = lo;
    lo = s0; hi = Nat;
    while (lo < hi) { int m = (lo + hi) >> 1; if (seg[m] < b + 1) lo = m + 1; else hi = m; }
    int e0 = lo;
    float sum = 0.f;
    for (int i = s0 + (int)threadIdx.x; i < e0; i += 256) sum += a[i];
    __shared__ float red[256];
    red[threadIdx.x] = sum; __syncthreads();
    for (int off = 128; off > 0; off >>= 1) {
        if ((int)threadIdx.x < off) red[threadIdx.x] += red[threadIdx.x + off];
        __syncthreads();
    }
    if (threadIdx.x == 0) anorm[b] = red[0];
}

// ---------- coef[n] = a[n] * |E[b]| / (anorm[b] + eps) ----------
__global__ void make_coef(const float* __restrict__ a, const float* __restrict__ anorm,
                          const float* __restrict__ E, const int* __restrict__ seg,
                          float* __restrict__ coef, int Nat) {
    int n = blockIdx.x * 256 + threadIdx.x;
    if (n >= Nat) return;
    int b = seg[n];
    coef[n] = a[n] * fabsf(E[b]) / (anorm[b] + EPS);
}

// ---------- WMMA fragment helpers (ISA 7.12.2 layouts, wave32) ----------
// A 16x32 bf16: lane m=lane&15; koff=(lane>>4)*8; elems 0..7 = K koff..koff+7,
// elems 8..15 = K koff+16..koff+23.
__device__ __forceinline__ v16bf load_afrag(const bf16_t* tbuf, int mt, int kc, int lane) {
    int m = lane & 15, koff = (lane >> 4) * 8;
    const bf16_t* row = tbuf + (mt * 16 + m) * TP + kc * 32 + koff;
    v8bf lo = *(const v8bf*)row;
    v8bf hi = *(const v8bf*)(row + 16);
    v16bf a;
    #pragma unroll
    for (int i = 0; i < 8; ++i) { a[i] = lo[i]; a[i + 8] = hi[i]; }
    return a;
}
// B 32x16 bf16: lane n=lane&15; kb=(lane>>4)*8; elems 0..7 = K kb..kb+7, elems 8..15 = K kb+16..kb+23.
// B[k][n] = W[nt*16+n][k] (out = in @ W^T).
__device__ __forceinline__ v16bf load_bfrag(const bf16_t* W, int nt, int kc, int lane) {
    int n = lane & 15, kb = (lane >> 4) * 8;
    const bf16_t* row = W + (nt * 16 + n) * F + kc * 32 + kb;
    v8bf lo = *(const v8bf*)row;
    v8bf hi = *(const v8bf*)(row + 16);
    v16bf b;
    #pragma unroll
    for (int i = 0; i < 8; ++i) { b[i] = lo[i]; b[i + 8] = hi[i]; }
    return b;
}

// One GEMM stage: this wave's 16-atom M-tile (mt) x all 16 N-tiles, K=256.
// N-tiles processed in 4 chunks of 4 -> only 4 live accumulators (32 VGPRs);
// chunk results go straight to LDS/global (no registers held across barriers).
// MODE 0: dst_t = bf16(silu(acc));  MODE 1: hbuf += acc;  MODE 2: out = acc.
template <int MODE>
__device__ __forceinline__ void gemm_stage(const bf16_t* __restrict__ tin,
                                           const bf16_t* __restrict__ W,
                                           bf16_t* __restrict__ dst_t,
                                           float* __restrict__ hbuf,
                                           float* __restrict__ out,
                                           size_t base, int Nat, int lane, int mt) {
    const int ml = lane & 15;          // C/D: col within tile
    const int mh = (lane >> 4) * 8;    // C/D: row = vgpr + 8*(lane>>4)
    for (int ch = 0; ch < 4; ++ch) {
        v8f acc[4];
        #pragma unroll
        for (int j = 0; j < 4; ++j)
            #pragma unroll
            for (int v = 0; v < 8; ++v) acc[j][v] = 0.f;
        for (int kc = 0; kc < 8; ++kc) {
            v16bf a = load_afrag(tin, mt, kc, lane);
            #pragma unroll
            for (int j = 0; j < 4; ++j) {
                v16bf b = load_bfrag(W, ch * 4 + j, kc, lane);
                acc[j] = __builtin_amdgcn_wmma_f32_16x16x32_bf16(
                    false, a, false, b, (short)0, acc[j], false, false);
            }
        }
        #pragma unroll
        for (int j = 0; j < 4; ++j) {
            const int nt = ch * 4 + j;
            #pragma unroll
            for (int v = 0; v < 8; ++v) {
                const int row = mt * 16 + mh + v;
                const int col = nt * 16 + ml;
                if (MODE == 0) {
                    dst_t[row * TP + col] = f2bf(silu_f(acc[j][v]));
                } else if (MODE == 1) {
                    hbuf[row * HP + col] += acc[j][v];
                } else {
                    size_t atom = base + (size_t)row;
                    if (atom < (size_t)Nat) out[atom * F + col] = acc[j][v];
                }
            }
        }
    }
}

// ---------- fused residual MLP: all 5 GEMMs, h-tile resident in LDS ----------
__global__ void __launch_bounds__(256, 1)
fused_resmlp(const float* __restrict__ coef, const float* __restrict__ Wv,
             const bf16_t* __restrict__ W1, const bf16_t* __restrict__ W2,
             const bf16_t* __restrict__ Wout,
             float* __restrict__ out, int Nat) {
    extern __shared__ char smem[];
    float*  hbuf = (float*)smem;                                   // [MTILE][HP] fp32
    bf16_t* tA   = (bf16_t*)(smem + MTILE * HP * sizeof(float));   // [MTILE][TP] bf16
    bf16_t* tB   = tA + MTILE * TP;                                // [MTILE][TP] bf16
    const int tid  = threadIdx.x;
    const int lane = tid & 31;
    const int mt   = tid >> 5;                    // wave id == atom sub-tile
    const size_t base = (size_t)blockIdx.x * MTILE;

    // h0[r][c] = coef[atom] * Wv[c]   (rank-1 init, fp32)
    for (int i = tid; i < MTILE * (F / 4); i += 256) {
        int r = i >> 6, c = (i & 63) << 2;
        size_t atom = base + (size_t)r;
        float cf = (atom < (size_t)Nat) ? coef[atom] : 0.f;
        float4 w = *(const float4*)&Wv[c];
        float4 h; h.x = cf * w.x; h.y = cf * w.y; h.z = cf * w.z; h.w = cf * w.w;
        *(float4*)&hbuf[r * HP + c] = h;
    }
    __syncthreads();

    for (int rb = 0; rb < 2; ++rb) {
        // tA = bf16(silu(h))
        for (int i = tid; i < MTILE * (F / 4); i += 256) {
            int r = i >> 6, c = (i & 63) << 2;
            float4 h4 = *(const float4*)&hbuf[r * HP + c];
            v4bf t;
            t[0] = f2bf(silu_f(h4.x)); t[1] = f2bf(silu_f(h4.y));
            t[2] = f2bf(silu_f(h4.z)); t[3] = f2bf(silu_f(h4.w));
            *(v4bf*)&tA[r * TP + c] = t;
        }
        __syncthreads();
        // tB = bf16(silu( tA @ W1^T ))
        gemm_stage<0>(tA, W1 + rb * F * F, tB, hbuf, out, base, Nat, lane, mt);
        __syncthreads();
        // h += tB @ W2^T
        gemm_stage<1>(tB, W2 + rb * F * F, tA, hbuf, out, base, Nat, lane, mt);
        __syncthreads();
    }
    // tA = bf16(silu(h))
    for (int i = tid; i < MTILE * (F / 4); i += 256) {
        int r = i >> 6, c = (i & 63) << 2;
        float4 h4 = *(const float4*)&hbuf[r * HP + c];
        v4bf t;
        t[0] = f2bf(silu_f(h4.x)); t[1] = f2bf(silu_f(h4.y));
        t[2] = f2bf(silu_f(h4.z)); t[3] = f2bf(silu_f(h4.w));
        *(v4bf*)&tA[r * TP + c] = t;
    }
    __syncthreads();
    // out = tA @ Wout^T
    gemm_stage<2>(tA, Wout, tB, hbuf, out, base, Nat, lane, mt);
}

// ---------- host ----------
extern "C" void kernel_launch(void* const* d_in, const int* in_sizes, int n_in,
                              void* d_out, int out_size, void* d_ws, size_t ws_size,
                              hipStream_t stream) {
    const float* x   = (const float*)d_in[0];
    const float* E   = (const float*)d_in[1];
    const int*   seg = (const int*)d_in[3];
    const float* Wq  = (const float*)d_in[4];
    const float* bq  = (const float*)d_in[5];
    const float* Wk  = (const float*)d_in[6];
    const float* Wv  = (const float*)d_in[7];
    const float* Wr1 = (const float*)d_in[8];
    const float* Wr2 = (const float*)d_in[9];
    const float* Wo  = (const float*)d_in[10];
    const int Bn  = in_sizes[1];
    const int Nat = in_sizes[3];
    const int NatP = (Nat + 3) & ~3;   // keep 16B alignment downstream

    // scratch layout in d_ws (~2.3 MB total)
    float* ws    = (float*)d_ws;
    float* anorm = ws;                  // [<=2048]
    float* a_arr = ws + 2048;           // [Nat]
    float* coef  = a_arr + NatP;        // [Nat]
    float* wkq   = coef + NatP;         // [256]
    float* kb    = wkq + 256;           // [1]
    bf16_t* W1b  = (bf16_t*)(wkq + 272);        // [2*256*256]
    bf16_t* W2b  = W1b + 2 * F * F;             // [2*256*256]
    bf16_t* Wob  = W2b + 2 * F * F;             // [256*256]

    make_wkq<<<1, 256, 0, stream>>>(Wq, bq, Wk, wkq, kb);
    cvt_weights<<<(2 * F * F + 255) / 256, 256, 0, stream>>>(Wr1, Wr2, Wo, W1b, W2b, Wob);
    dot_softplus<<<(Nat + 7) / 8, 256, 0, stream>>>(x, E, seg, wkq, kb, a_arr, Nat);
    seg_reduce<<<Bn, 256, 0, stream>>>(a_arr, seg, anorm, Nat);
    make_coef<<<(Nat + 255) / 256, 256, 0, stream>>>(a_arr, anorm, E, seg, coef, Nat);

    const size_t shmem = (size_t)MTILE * HP * sizeof(float)
                       + 2 * (size_t)MTILE * TP * sizeof(bf16_t);
    hipFuncSetAttribute((const void*)fused_resmlp,
                        hipFuncAttributeMaxDynamicSharedMemorySize, (int)shmem);
    fused_resmlp<<<(Nat + MTILE - 1) / MTILE, 256, shmem, stream>>>(
        coef, Wv, W1b, W2b, Wob, (float*)d_out, Nat);
}
